// GraphAttentionLayer_43404939493908
// MI455X (gfx1250) — matrix operand order repaired and checked by reading, not compiled, not run on hardware
//
#include <hip/hip_runtime.h>

// ---------------------------------------------------------------------------
// GAT layer for MI455X (gfx1250, wave32, WMMA + async-to-LDS staging).
//   B=8, N=1024, F_IN=64, HEADS=8, D=64, C=H*D=512
// Math: softmax over axis=1 (i)  =>  attn[i,j] = E[i,j]/denom[j],
//   E[i,j]=exp(lrelu(a_src[i]+a_dst[j])),  denom[j]=sum_i E[i,j]
//   h' = E @ (rdenom ⊙ Wh)   -- fold rdenom into a bf16 Wh' (transposed).
// All three GEMMs use v_wmma_f32_16x16x32_bf16; E tiles built in registers.
// Dominant GEMM stages its shared B tiles via global_load_async_to_lds_b128
// (double-buffered, s_wait_asynccnt) so 8 waves share one copy per block.
// ---------------------------------------------------------------------------

#define B_   8
#define N_   1024
#define FIN  64
#define H_   8
#define DH   64
#define C_   512   // H_*DH

typedef __bf16 bf16_t;
typedef __attribute__((ext_vector_type(16))) __bf16 v16bf;
typedef __attribute__((ext_vector_type(8)))  float  v8f;

struct alignas(16) F4 { float x, y, z, w; };
union Frag { v16bf v; F4 f4[2]; bf16_t e[16]; };

static __device__ inline v8f wmma_bf16(v16bf a, v16bf b, v8f c) {
  // (neg_a, A, neg_b, B, c_mod, C, reuse_a, reuse_b)
  return __builtin_amdgcn_wmma_f32_16x16x32_bf16(false, a, false, b,
                                                 (short)0, c, false, false);
}

// ---------------------------------------------------------------------------
// K0: bf16 copies: h (same layout), W^T [512][64], out_W^T [64][512]
// ---------------------------------------------------------------------------
__global__ void k_prep(const float* __restrict__ h, const float* __restrict__ W,
                       const float* __restrict__ outW,
                       bf16_t* __restrict__ h_bf, bf16_t* __restrict__ w_t,
                       bf16_t* __restrict__ outw_t) {
  int e = blockIdx.x * 256 + threadIdx.x;
  if (e < B_ * N_ * FIN) { h_bf[e] = (bf16_t)h[e]; return; }
  e -= B_ * N_ * FIN;
  if (e < C_ * FIN) {               // W^T[d][k] = W[k][d]
    int d = e / FIN, k = e % FIN;
    w_t[e] = (bf16_t)W[k * C_ + d];
    return;
  }
  e -= C_ * FIN;
  if (e < FIN * C_) {               // outW^T[d][k] = out_W[k][d]
    int d = e / C_, k = e % C_;
    outw_t[e] = (bf16_t)outW[k * FIN + d];
    return;
  }
}

// ---------------------------------------------------------------------------
// K1: Wh = h @ W  (bf16 WMMA, K=64).  64 blocks x 256 threads.
// Block = (b, 128-row tile). Wave w -> rows [w*16, w*16+16), loops heads.
// ---------------------------------------------------------------------------
__global__ void k_wh(const bf16_t* __restrict__ h_bf,
                     const bf16_t* __restrict__ w_t,
                     bf16_t* __restrict__ wh_bf) {
  int blk = blockIdx.x;
  int b = blk >> 3, it = blk & 7;
  int t = threadIdx.x, w = t >> 5, lane = t & 31, m = lane & 15;
  bool hi = lane >= 16;
  int i = it * 128 + w * 16 + m;

  const bf16_t* arow = h_bf + ((size_t)b * N_ + i) * FIN;
  int ko = hi ? 16 : 0;   // A fragment K base (contiguous 16-run per half)
  int kb = hi ? 16 : 0;   // B fragment K base

  Frag a0, a1;
  a0.f4[0] = *(const F4*)(arow + ko);
  a0.f4[1] = *(const F4*)(arow + ko + 8);
  a1.f4[0] = *(const F4*)(arow + 32 + ko);
  a1.f4[1] = *(const F4*)(arow + 32 + ko + 8);

  v8f zero = {0.f, 0.f, 0.f, 0.f, 0.f, 0.f, 0.f, 0.f};

  for (int hh = 0; hh < H_; ++hh) {
#pragma unroll
    for (int nt = 0; nt < 4; ++nt) {
      int col = hh * DH + nt * 16 + m;               // output column (N index)
      const bf16_t* bcol = w_t + (size_t)col * FIN;  // W^T row: k contiguous
      Frag b0, b1;
      b0.f4[0] = *(const F4*)(bcol + kb);
      b0.f4[1] = *(const F4*)(bcol + kb + 8);
      b1.f4[0] = *(const F4*)(bcol + 32 + kb);
      b1.f4[1] = *(const F4*)(bcol + 32 + kb + 8);
      v8f acc = zero;
      acc = wmma_bf16(a0.v, b0.v, acc);
      acc = wmma_bf16(a1.v, b1.v, acc);
#pragma unroll
      for (int r = 0; r < 8; ++r) {
        int mr = hi ? r + 8 : r;   // C layout: VGPR r -> M=r (lo) / r+8 (hi)
        wh_bf[((size_t)b * N_ + it * 128 + w * 16 + mr) * C_ + col] =
            (bf16_t)acc[r];
      }
    }
  }
}

// ---------------------------------------------------------------------------
// K2: a_src[b,n,h] = Wh[b,n,h,:].a[:64];  a_dst = Wh . a[64:]
// ---------------------------------------------------------------------------
__global__ void k_coef(const bf16_t* __restrict__ wh_bf,
                       const float* __restrict__ a,
                       float* __restrict__ a_src, float* __restrict__ a_dst) {
  int idx = blockIdx.x * 256 + threadIdx.x;  // ((b*N+n)*H + hh)
  int hh = idx & 7;
  int bn = idx >> 3;
  const bf16_t* row = wh_bf + (size_t)bn * C_ + hh * DH;
  float s0 = 0.f, s1 = 0.f;
#pragma unroll 8
  for (int d = 0; d < DH; ++d) {
    float wv = (float)row[d];
    s0 += wv * a[d];
    s1 += wv * a[DH + d];
  }
  a_src[idx] = s0;
  a_dst[idx] = s1;
}

// ---------------------------------------------------------------------------
// K3: rdenom[b,h,j] = 1 / sum_i exp(lrelu(s_i + t_j)).
// Logits are O(0.1) (stdv-scaled weights) -> exp is safe without max-shift.
// Grid: B*H*4 blocks, thread = one j, i staged in LDS.
// ---------------------------------------------------------------------------
__global__ void k_denom(const float* __restrict__ a_src,
                        const float* __restrict__ a_dst,
                        float* __restrict__ rdenom) {
  __shared__ float s[N_];
  int blk = blockIdx.x;
  int b = blk >> 5, hh = (blk >> 2) & 7, j0 = (blk & 3) * 256;
  int t = threadIdx.x;
#pragma unroll
  for (int q = 0; q < 4; ++q) {
    int i = q * 256 + t;
    s[i] = a_src[((size_t)b * N_ + i) * H_ + hh];
  }
  __syncthreads();
  int j = j0 + t;
  float tj = a_dst[((size_t)b * N_ + j) * H_ + hh];
  float sum = 0.f;
  for (int i = 0; i < N_; ++i) {
    float z = s[i] + tj;
    z = fmaxf(z, 0.01f * z);  // leaky_relu
    sum += __expf(z);
  }
  rdenom[((size_t)b * H_ + hh) * N_ + j] = 1.0f / sum;
}

// ---------------------------------------------------------------------------
// K4: Wh'[b][h][d][j] = bf16(rdenom[j] * Wh[b,j,h,d])  (LDS transpose).
// j-contiguous rows -> WMMA B fragments become contiguous 16-element runs.
// ---------------------------------------------------------------------------
__global__ void k_whp(const bf16_t* __restrict__ wh_bf,
                      const float* __restrict__ rdenom,
                      bf16_t* __restrict__ whp) {
  __shared__ bf16_t tile[256 * 65];  // pitch 65 -> conflict-free transpose
  int blk = blockIdx.x;
  int b = blk >> 5, hh = (blk >> 2) & 7, j0 = (blk & 3) * 256;
  int t = threadIdx.x;
  const float* rd = rdenom + ((size_t)b * H_ + hh) * N_ + j0;
  for (int k = 0; k < 64; ++k) {      // coalesced read (d contiguous)
    int e = k * 256 + t;
    int jl = e >> 6, d = e & 63;
    float v = (float)wh_bf[((size_t)b * N_ + j0 + jl) * C_ + hh * DH + d] *
              rd[jl];
    tile[jl * 65 + d] = (bf16_t)v;
  }
  __syncthreads();
  bf16_t* outb = whp + (size_t)(b * H_ + hh) * DH * N_ + j0;
  for (int k = 0; k < 64; ++k)        // coalesced write (j contiguous)
    outb[(size_t)k * N_ + t] = tile[t * 65 + k];
}

// ---------------------------------------------------------------------------
// K5 (dominant, 8.6 GF): h'[b,h] = E @ Wh'   with E built in registers.
// 512 blocks = (b,h,128-row tile); wave = 16 rows x 64 cols, K=1024.
// B tiles (shared by all 8 waves) staged once per block per K-step through
// LDS with global_load_async_to_lds_b128, double-buffered + s_wait_asynccnt.
// Per K-step/lane: 16 v_exp_f32 (A fragment) + 4 WMMA + 8 ds_load_b128.
// ---------------------------------------------------------------------------
__global__ void k_attn(const float* __restrict__ a_src,
                       const float* __restrict__ a_dst,
                       const bf16_t* __restrict__ whp,
                       bf16_t* __restrict__ hp_bf) {
  __shared__ float sd[N_];                       // a_dst slice for this (b,h)
  __shared__ __align__(16) bf16_t bt[2][64][72]; // double-buffered B tile,
                                                 // pitch 72 (anti-conflict)
  int blk = blockIdx.x;
  int b = blk >> 6, hh = (blk >> 3) & 7, it = blk & 7;
  int i0 = it * 128;
  int t = threadIdx.x, w = t >> 5, lane = t & 31, m = lane & 15;
  bool hi = lane >= 16;
#pragma unroll
  for (int q = 0; q < 4; ++q) {
    int j = q * 256 + t;
    sd[j] = a_dst[((size_t)b * N_ + j) * H_ + hh];
  }

  int i = i0 + w * 16 + m;
  float sm = a_src[((size_t)b * N_ + i) * H_ + hh];
  const bf16_t* wbase = whp + (size_t)(b * H_ + hh) * DH * N_;

  // Async stage of one 64x32 B tile (4 KB): 1x b128 per thread.
  int sd_row = t >> 2;            // d = 0..63
  int sd_col = (t & 3) * 8;       // 8 bf16 = 16 B chunk
  const bf16_t* gsrc0 = wbase + (size_t)sd_row * N_ + sd_col;
#define STAGE(buf, jb)                                                        \
  do {                                                                        \
    unsigned _lds = (unsigned)(uintptr_t)(&bt[(buf)][sd_row][sd_col]);        \
    const bf16_t* _g = gsrc0 + (jb);                                          \
    asm volatile("global_load_async_to_lds_b128 %0, %1, off"                  \
                 :: "v"(_lds), "v"(_g) : "memory");                           \
  } while (0)

  STAGE(0, 0);  // prologue: tile for jb=0

  v8f zero = {0.f, 0.f, 0.f, 0.f, 0.f, 0.f, 0.f, 0.f};
  v8f acc[4] = {zero, zero, zero, zero};
  int ko = hi ? 16 : 0;
  int kb = hi ? 16 : 0;

  int s = 0;
  for (int jb = 0; jb < N_; jb += 32, s ^= 1) {
    // Own copies of bt[s] done; barrier => everyone's copies visible AND
    // all readers of bt[s^1] (previous step) finished, safe to overwrite.
    asm volatile("s_wait_asynccnt 0x0" ::: "memory");
    __syncthreads();
    if (jb + 32 < N_) STAGE(s ^ 1, jb + 32);   // overlap copy with compute

    // --- A fragment: E[i, jb..jb+31] in contiguous k order (k = 16*half + s, same as B) ---
    F4 t0 = *(const F4*)(&sd[jb + ko]);
    F4 t1 = *(const F4*)(&sd[jb + ko + 4]);
    F4 t2 = *(const F4*)(&sd[jb + ko + 8]);
    F4 t3 = *(const F4*)(&sd[jb + ko + 12]);
    Frag af;
#define LREXP(x) ((bf16_t)__expf(fmaxf((sm + (x)), 0.01f * (sm + (x)))))
    af.e[0]  = LREXP(t0.x); af.e[1]  = LREXP(t0.y);
    af.e[2]  = LREXP(t0.z); af.e[3]  = LREXP(t0.w);
    af.e[4]  = LREXP(t1.x); af.e[5]  = LREXP(t1.y);
    af.e[6]  = LREXP(t1.z); af.e[7]  = LREXP(t1.w);
    af.e[8]  = LREXP(t2.x); af.e[9]  = LREXP(t2.y);
    af.e[10] = LREXP(t2.z); af.e[11] = LREXP(t2.w);
    af.e[12] = LREXP(t3.x); af.e[13] = LREXP(t3.y);
    af.e[14] = LREXP(t3.z); af.e[15] = LREXP(t3.w);
#undef LREXP

    // --- B fragments from the staged LDS tile: 2x ds_load_b128 each ---
#pragma unroll
    for (int nt = 0; nt < 4; ++nt) {
      const bf16_t* bp = &bt[s][nt * 16 + m][kb];
      Frag bf;
      bf.f4[0] = *(const F4*)(bp);
      bf.f4[1] = *(const F4*)(bp + 8);
      acc[nt] = wmma_bf16(af.v, bf.v, acc[nt]);
    }
  }
#undef STAGE

#pragma unroll
  for (int nt = 0; nt < 4; ++nt)
#pragma unroll
    for (int r = 0; r < 8; ++r) {
      int mr = hi ? r + 8 : r;
      hp_bf[((size_t)b * N_ + i0 + w * 16 + mr) * C_ + hh * DH + nt * 16 + m] =
          (bf16_t)acc[nt][r];
    }
}

// ---------------------------------------------------------------------------
// K6: out = h' @ out_W + out_b (bf16 WMMA, K=512), fp32 output.
// ---------------------------------------------------------------------------
__global__ void k_out(const bf16_t* __restrict__ hp_bf,
                      const bf16_t* __restrict__ outw_t,
                      const float* __restrict__ out_b,
                      float* __restrict__ out) {
  int blk = blockIdx.x;  // 64 blocks
  int b = blk >> 3, it = blk & 7;
  int t = threadIdx.x, w = t >> 5, lane = t & 31, m = lane & 15;
  bool hi = lane >= 16;
  int i = it * 128 + w * 16 + m;
  const bf16_t* arow = hp_bf + ((size_t)b * N_ + i) * C_;
  int ko = hi ? 16 : 0, kb = hi ? 16 : 0;

  v8f zero = {0.f, 0.f, 0.f, 0.f, 0.f, 0.f, 0.f, 0.f};
  v8f acc[4] = {zero, zero, zero, zero};

  for (int ks = 0; ks < C_; ks += 32) {
    Frag af;
    af.f4[0] = *(const F4*)(arow + ks + ko);
    af.f4[1] = *(const F4*)(arow + ks + ko + 8);
#pragma unroll
    for (int nt = 0; nt < 4; ++nt) {
      const bf16_t* bp = outw_t + (size_t)(nt * 16 + m) * C_ + ks + kb;
      Frag bf;
      bf.f4[0] = *(const F4*)(bp);
      bf.f4[1] = *(const F4*)(bp + 8);
      acc[nt] = wmma_bf16(af.v, bf.v, acc[nt]);
    }
  }

#pragma unroll
  for (int nt = 0; nt < 4; ++nt) {
    int d = nt * 16 + m;
    float bias = out_b[d];
#pragma unroll
    for (int r = 0; r < 8; ++r) {
      int mr = hi ? r + 8 : r;
      out[((size_t)b * N_ + it * 128 + w * 16 + mr) * FIN + d] =
          acc[nt][r] + bias;
    }
  }
}

// ---------------------------------------------------------------------------
extern "C" void kernel_launch(void* const* d_in, const int* in_sizes, int n_in,
                              void* d_out, int out_size, void* d_ws,
                              size_t ws_size, hipStream_t stream) {
  (void)in_sizes; (void)n_in; (void)out_size; (void)ws_size;
  const float* h    = (const float*)d_in[0];
  // d_in[1] = adj (unused by the module)
  const float* W    = (const float*)d_in[2];
  const float* a    = (const float*)d_in[3];
  const float* outW = (const float*)d_in[4];
  const float* outb = (const float*)d_in[5];
  float* out = (float*)d_out;

  char* ws = (char*)d_ws;
  size_t off = 0;
  auto alloc = [&](size_t bytes) -> void* {
    void* p = ws + off;
    off += (bytes + 255) & ~(size_t)255;
    return p;
  };
  bf16_t* h_bf   = (bf16_t*)alloc((size_t)B_ * N_ * FIN * 2);     // 1 MB
  bf16_t* w_t    = (bf16_t*)alloc((size_t)C_ * FIN * 2);          // 64 KB
  bf16_t* outw_t = (bf16_t*)alloc((size_t)FIN * C_ * 2);          // 64 KB
  bf16_t* wh_bf  = (bf16_t*)alloc((size_t)B_ * N_ * C_ * 2);      // 8 MB
  float*  a_src  = (float*) alloc((size_t)B_ * N_ * H_ * 4);      // 256 KB
  float*  a_dst  = (float*) alloc((size_t)B_ * N_ * H_ * 4);      // 256 KB
  float*  rden   = (float*) alloc((size_t)B_ * H_ * N_ * 4);      // 256 KB
  bf16_t* whp    = (bf16_t*)alloc((size_t)B_ * H_ * DH * N_ * 2); // 8 MB
  bf16_t* hp_bf  = (bf16_t*)alloc((size_t)B_ * N_ * C_ * 2);      // 8 MB

  int prepElems = B_ * N_ * FIN + C_ * FIN + FIN * C_;
  k_prep<<<(prepElems + 255) / 256, 256, 0, stream>>>(h, W, outW, h_bf, w_t,
                                                      outw_t);
  k_wh<<<B_ * (N_ / 128), 256, 0, stream>>>(h_bf, w_t, wh_bf);
  k_coef<<<(B_ * N_ * H_) / 256, 256, 0, stream>>>(wh_bf, a, a_src, a_dst);
  k_denom<<<B_ * H_ * 4, 256, 0, stream>>>(a_src, a_dst, rden);
  k_whp<<<B_ * H_ * 4, 256, 0, stream>>>(wh_bf, rden, whp);
  k_attn<<<B_ * H_ * (N_ / 128), 256, 0, stream>>>(a_src, a_dst, whp, hp_bf);
  k_out<<<B_ * (N_ / 128), 256, 0, stream>>>(hp_bf, outw_t, outb, out);
}
